// SelfAttention_21088289423838
// MI455X (gfx1250) — matrix operand order repaired and checked by reading, not compiled, not run on hardware
//
#include <hip/hip_runtime.h>
#include <hip/hip_bf16.h>

// ---------------------------------------------------------------------------
// Self-attention forward, CDNA5 (gfx1250), wave32, WMMA f16->f32.
// B=2, S=2048, E=1024, H=16, HD=64.
// v5: async global->LDS staging (ASYNCcnt) with race-free issue ordering
//     (wait -> barrier -> issue-next-DMA -> compute), A-frag prefetch issued
//     after the barrier, and batched B-fragment LDS loads before WMMA groups.
// ---------------------------------------------------------------------------

typedef _Float16 v16h __attribute__((ext_vector_type(16)));
typedef _Float16 v8h  __attribute__((ext_vector_type(8)));
typedef _Float16 v4h  __attribute__((ext_vector_type(4)));
typedef float    v8f  __attribute__((ext_vector_type(8)));

#define WMMA(A, B, C) \
  __builtin_amdgcn_wmma_f32_16x16x32_f16(false, (A), false, (B), (short)0, (C), false, false)

union V16U { v16h v; v8h h[2]; };

#if defined(__has_builtin)
#  if __has_builtin(__builtin_amdgcn_global_load_async_to_lds_b128)
#    define HAVE_ASYNC_LDS 1
#  endif
#endif

#ifdef HAVE_ASYNC_LDS
#  if __has_builtin(__builtin_amdgcn_s_wait_asynccnt)
#    define ASYNC_WAIT(n) __builtin_amdgcn_s_wait_asynccnt(n)
#  else
#    define ASYNC_WAIT(n) asm volatile("s_wait_asynccnt %0" ::"i"(n))
#  endif
// Builtin signature (from compiler diagnostic): first param is
// `int __attribute__((vector_size(16))) __device__ *` i.e. int4 in AS(1).
typedef int v4i_t __attribute__((vector_size(16)));
typedef __attribute__((address_space(1))) v4i_t* as1_v4i;
typedef __attribute__((address_space(3))) v4i_t* as3_v4i;

// Copy 32B (per lane) global -> LDS without touching VGPR data paths.
// Generic->AS(1): same address value.  Generic->AS(3): low 32 bits of a
// generic LDS address are the LDS byte offset (flat aperture rule).
static __device__ __forceinline__ void async_cp32B(const _Float16* g,
                                                   _Float16* l) {
  as1_v4i gp = (as1_v4i)(uintptr_t)g;
  as3_v4i lp = (as3_v4i)(unsigned int)(uintptr_t)l;
  __builtin_amdgcn_global_load_async_to_lds_b128(gp, lp, 0, 0);
  __builtin_amdgcn_global_load_async_to_lds_b128(gp, lp, 16, 0);
}
#endif

// A-fragment (16x32 f16, MxK): lane row = lane&15 (baked into `base`),
// lanes 0-15 hold K=0..15, lanes 16-31 hold K=16..31 (contiguous in K).
static __device__ __forceinline__ v16h load_a_frag(const _Float16* base, int g) {
  V16U u;
  u.h[0] = *(const v8h*)(base + 16 * g);
  u.h[1] = *(const v8h*)(base + 16 * g + 8);
  return u.v;
}

// B-fragment (32x16 f16, KxN): lane col = lane&15 (baked into `base`),
// lanes 0-15 hold K=0..15, lanes 16-31 hold K=16..31 (contiguous in K).
static __device__ __forceinline__ v16h load_b_frag(const _Float16* base, int g) {
  return *(const v16h*)(base + 16 * g);
}

// ---------------------------------------------------------------------------
// fp32 -> fp16 conversion, 4 elements/thread
// ---------------------------------------------------------------------------
__global__ void f32_to_f16_kernel(const float* __restrict__ src,
                                  _Float16* __restrict__ dst, int n) {
  int i = (blockIdx.x * blockDim.x + threadIdx.x) * 4;
  if (i + 3 < n) {
    float4 f = *(const float4*)(src + i);
    v4h o;
    o[0] = (_Float16)f.x; o[1] = (_Float16)f.y;
    o[2] = (_Float16)f.z; o[3] = (_Float16)f.w;
    *(v4h*)(dst + i) = o;
  }
}

// ---------------------------------------------------------------------------
// Fused QKV projection: y = x @ W^T.  x:[4096,1024]f16, W:[1024,1024]f16.
// blockIdx.z selects Q/K/V.  Q scaled by 1/sqrt(64).
// W k-tiles async-staged in LDS (shared by all 4 waves), double-buffered.
// Each wave: 16 rows x 64 cols (4 WMMA col-tiles), K-loop in steps of 32.
// ---------------------------------------------------------------------------
__global__ __launch_bounds__(128) void qkv_proj_kernel(
    const _Float16* __restrict__ xh,
    const _Float16* __restrict__ Wqh,
    const _Float16* __restrict__ Wkh,
    const _Float16* __restrict__ Wvh,
    _Float16* __restrict__ Qh,
    _Float16* __restrict__ Kh,
    _Float16* __restrict__ Vt) {
  __shared__ __align__(16) _Float16 LW[2][64 * 32];  // [ncol][k] tile, 4KB x2

  const int t    = threadIdx.x;
  const int lane = t & 31;
  const int wave = t >> 5;
  const int g    = lane >> 4;
  const int ln   = lane & 15;
  const int which = blockIdx.z;
  const _Float16* W = (which == 0) ? Wqh : ((which == 1) ? Wkh : Wvh);

  const int m0 = blockIdx.y * 64 + wave * 16;  // token-row tile base
  const int n0 = blockIdx.x * 64;              // output-col tile base

  // cooperative W-tile copy: thread t moves 32B (row = t/2, k-half = 16*(t&1))
  const int wrow  = t >> 1;
  const int whalf = (t & 1) * 16;
  const _Float16* wsrc = W + (n0 + wrow) * 1024 + whalf;
  _Float16* wdst0 = &LW[0][wrow * 32 + whalf];
  _Float16* wdst1 = &LW[1][wrow * 32 + whalf];

  const _Float16* arow = xh + (m0 + ln) * 1024;

  // prologue: stage k-chunk 0
#ifdef HAVE_ASYNC_LDS
  async_cp32B(wsrc, wdst0);
#else
  *(v16h*)wdst0 = *(const v16h*)(wsrc);
#endif
  v16h a = load_a_frag(arow, g);

  v8f acc[4] = {};
  int cur = 0;
  for (int k0 = 0; k0 < 1024; k0 += 32) {
    const bool more = (k0 + 32) < 1024;
    v16h anext = {};
#ifdef HAVE_ASYNC_LDS
    ASYNC_WAIT(0);    // LW[cur] DMA (issued last iteration) has landed
    __syncthreads();  // all waves' DMA landed; LW[cur^1] readers done
    if (more) {       // DMA next buffer behind this iteration's WMMAs
      async_cp32B(wsrc + k0 + 32, cur ? wdst0 : wdst1);
      anext = load_a_frag(arow + k0 + 32, g);
    }
#else
    __syncthreads();  // LW[cur] committed by all waves
    v16h wnext = {};
    if (more) {
      wnext = *(const v16h*)(wsrc + k0 + 32);
      anext = load_a_frag(arow + k0 + 32, g);
    }
#endif
    v16h bf[4];
#pragma unroll
    for (int nt = 0; nt < 4; nt++) {
      bf[nt] = *(const v16h*)&LW[cur][(nt * 16 + ln) * 32 + 16 * g];
    }
#pragma unroll
    for (int nt = 0; nt < 4; nt++) {
      acc[nt] = WMMA(a, bf[nt], acc[nt]);
    }
#ifndef HAVE_ASYNC_LDS
    if (more) *(v16h*)(cur ? wdst0 : wdst1) = wnext;
#endif
    if (more) a = anext;
    cur ^= 1;
  }

  const float qscale = (which == 0) ? 0.125f : 1.0f;  // 1/sqrt(HD) on Q
#pragma unroll
  for (int nt = 0; nt < 4; nt++) {
    const int ncol = n0 + nt * 16 + ln;
    const int h = ncol >> 6, d = ncol & 63;
#pragma unroll
    for (int r = 0; r < 8; r++) {
      const int mrow = m0 + r + 8 * g;          // D rows: m = r + 8*(lane/16)
      const int b = mrow >> 11, s = mrow & 2047;
      const float v = (float)acc[nt][r];
      if (which == 2) {
        Vt[((b * 16 + h) * 64 + d) * 2048 + s] = (_Float16)v;       // [B,H,HD,S]
      } else {
        _Float16* dst = (which == 0) ? Qh : Kh;
        dst[((b * 16 + h) * 2048 + s) * 64 + d] = (_Float16)(v * qscale);
      }
    }
  }
}

// ---------------------------------------------------------------------------
// Flash attention: block = 64 queries of one (b,h); wave = 16 queries.
// K/V blocks (32 keys) async-staged in LDS, double-buffered; per key block:
// QK^T (4 WMMA), online softmax, P->LDS layout fixup (per-wave region,
// LDS is in-order per wave -> no barrier), P@V (4 WMMA).
// ---------------------------------------------------------------------------
__global__ __launch_bounds__(128) void flash_attn_kernel(
    const _Float16* __restrict__ Qh,
    const _Float16* __restrict__ Kh,
    const _Float16* __restrict__ Vt,
    _Float16* __restrict__ Ah) {
  __shared__ __align__(16) _Float16 LK[2][32 * 64];  // [key][d]   4KB x2
  __shared__ __align__(16) _Float16 LV[2][64 * 32];  // [d][key]   4KB x2
  __shared__ __align__(16) _Float16 Pl[4][16 * 32];  // per-wave P tile

  const int t    = threadIdx.x;
  const int lane = t & 31;
  const int wave = t >> 5;
  const int g    = lane >> 4;
  const int ln   = lane & 15;
  const int h = blockIdx.y, b = blockIdx.z;
  const int bh = b * 16 + h;
  const int q0 = blockIdx.x * 64 + wave * 16;

  const _Float16* Kbase = Kh + (size_t)bh * 2048 * 64;  // [key][d]
  const _Float16* Vbase = Vt + (size_t)bh * 64 * 2048;  // [d][key]

  // cooperative V copy: thread t moves 32B (d-row = t/2, key-half = 16*(t&1))
  const int vrow  = t >> 1;
  const int vhalf = (t & 1) * 16;
  _Float16* kdst[2] = { &LK[0][t * 16], &LK[1][t * 16] };
  _Float16* vdst[2] = { &LV[0][vrow * 32 + vhalf], &LV[1][vrow * 32 + vhalf] };

  // Q A-fragments for both 32-wide d-chunks (HD = 64)
  const _Float16* qrow = Qh + ((size_t)bh * 2048 + q0 + ln) * 64;
  const v16h qa0 = load_a_frag(qrow, g);
  const v16h qa1 = load_a_frag(qrow + 32, g);

  // prologue: stage key-block 0 (K block is one contiguous 4KB run)
#ifdef HAVE_ASYNC_LDS
  async_cp32B(Kbase + t * 16, kdst[0]);
  async_cp32B(Vbase + vrow * 2048 + vhalf, vdst[0]);
#else
  *(v16h*)kdst[0] = *(const v16h*)(Kbase + t * 16);
  *(v16h*)vdst[0] = *(const v16h*)(Vbase + vrow * 2048 + vhalf);
#endif

  float M[8], L[8];
  v8f O[4] = {};
#pragma unroll
  for (int r = 0; r < 8; r++) { M[r] = -__builtin_inff(); L[r] = 0.0f; }

  _Float16* P = Pl[wave];
  int cur = 0;

  for (int kb = 0; kb < 2048; kb += 32) {
    const bool more = (kb + 32) < 2048;
#ifdef HAVE_ASYNC_LDS
    ASYNC_WAIT(0);    // LK/LV[cur] DMA (issued last iteration) has landed
    __syncthreads();  // all waves' DMA landed; LK/LV[cur^1] readers done
    if (more) {       // DMA next key block behind this iteration's compute
      async_cp32B(Kbase + (kb + 32) * 64 + t * 16, kdst[cur ^ 1]);
      async_cp32B(Vbase + vrow * 2048 + kb + 32 + vhalf, vdst[cur ^ 1]);
    }
#else
    __syncthreads();  // LK/LV[cur] committed by all waves
    v16h knext = {}, vnext = {};
    if (more) {
      knext = *(const v16h*)(Kbase + (kb + 32) * 64 + t * 16);
      vnext = *(const v16h*)(Vbase + vrow * 2048 + kb + 32 + vhalf);
      __builtin_prefetch(Kbase + (kb + 64) * 64 + t * 16, 0, 1);
      __builtin_prefetch(Vbase + vrow * 2048 + kb + 64 + vhalf, 0, 1);
    }
#endif

    // ---- scores: two 16-key tiles, contraction over d (2 x WMMA each) ----
    const _Float16* kp0 = &LK[cur][ln * 64];         // tile0: key = ln
    const _Float16* kp1 = &LK[cur][(16 + ln) * 64];  // tile1: key = 16+ln
    v16h kf0 = load_b_frag(kp0, g);
    v16h kf1 = load_b_frag(kp0 + 32, g);
    v16h kf2 = load_b_frag(kp1, g);
    v16h kf3 = load_b_frag(kp1 + 32, g);
    v8f s0 = {}, s1 = {};
    s0 = WMMA(qa0, kf0, s0);
    s0 = WMMA(qa1, kf1, s0);
    s1 = WMMA(qa0, kf2, s1);
    s1 = WMMA(qa1, kf3, s1);

    // ---- online softmax over this 16x32 block ----
    float p0[8], p1[8];
#pragma unroll
    for (int r = 0; r < 8; r++) {
      float v = fmaxf((float)s0[r], (float)s1[r]);
      v = fmaxf(v, __shfl_xor(v, 1, 32));
      v = fmaxf(v, __shfl_xor(v, 2, 32));
      v = fmaxf(v, __shfl_xor(v, 4, 32));
      v = fmaxf(v, __shfl_xor(v, 8, 32));   // row max within 16-lane group
      const float mnew = fmaxf(M[r], v);
      const float a0 = __expf((float)s0[r] - mnew);
      const float a1 = __expf((float)s1[r] - mnew);
      p0[r] = a0; p1[r] = a1;
      float rs = a0 + a1;
      rs += __shfl_xor(rs, 1, 32);
      rs += __shfl_xor(rs, 2, 32);
      rs += __shfl_xor(rs, 4, 32);
      rs += __shfl_xor(rs, 8, 32);          // row sum
      const float alpha = __expf(M[r] - mnew);
      L[r] = alpha * L[r] + rs;
      M[r] = mnew;
#pragma unroll
      for (int dt = 0; dt < 4; dt++) O[dt][r] = O[dt][r] * alpha;
    }

    // ---- P tile: C-fragment -> A-fragment via per-wave LDS region.
    // LDS ops from one wave execute in order (DScnt in-order), so the
    // cross-lane store->load needs no block barrier.
#pragma unroll
    for (int r = 0; r < 8; r++) {
      P[(r + 8 * g) * 32 + ln]      = (_Float16)p0[r];
      P[(r + 8 * g) * 32 + 16 + ln] = (_Float16)p1[r];
    }
    V16U pa;
    pa.h[0] = *(const v8h*)&P[ln * 32 + 16 * g];
    pa.h[1] = *(const v8h*)&P[ln * 32 + 16 * g + 8];

    // ---- O += P @ V : contraction over 32 keys, 4 d-tiles ----
    v16h vf[4];
#pragma unroll
    for (int dt = 0; dt < 4; dt++) {
      vf[dt] = load_b_frag(&LV[cur][(dt * 16 + ln) * 32], g);
    }
#pragma unroll
    for (int dt = 0; dt < 4; dt++) {
      O[dt] = WMMA(pa.v, vf[dt], O[dt]);
    }

#ifndef HAVE_ASYNC_LDS
    if (more) {  // commit next block behind this iteration's compute
      *(v16h*)kdst[cur ^ 1] = knext;
      *(v16h*)vdst[cur ^ 1] = vnext;
    }
#endif
    cur ^= 1;
  }

  // ---- normalize and write concat-head output [B,S,E] (f16) ----
#pragma unroll
  for (int r = 0; r < 8; r++) {
    const float inv = 1.0f / L[r];
    const int q = q0 + r + 8 * g;
#pragma unroll
    for (int dt = 0; dt < 4; dt++) {
      Ah[(b * 2048 + q) * 1024 + h * 64 + dt * 16 + ln] =
          (_Float16)(O[dt][r] * inv);
    }
  }
}

// ---------------------------------------------------------------------------
// Output projection: out = Ah @ Wo^T, fp32 result to d_out.
// Same async-staged double-buffered structure as qkv_proj_kernel.
// ---------------------------------------------------------------------------
__global__ __launch_bounds__(128) void out_proj_kernel(
    const _Float16* __restrict__ Ah,
    const _Float16* __restrict__ Woh,
    float* __restrict__ out) {
  __shared__ __align__(16) _Float16 LW[2][64 * 32];

  const int t    = threadIdx.x;
  const int lane = t & 31;
  const int wave = t >> 5;
  const int g = lane >> 4, ln = lane & 15;
  const int m0 = blockIdx.y * 64 + wave * 16;
  const int n0 = blockIdx.x * 64;

  const int wrow  = t >> 1;
  const int whalf = (t & 1) * 16;
  const _Float16* wsrc = Woh + (n0 + wrow) * 1024 + whalf;
  _Float16* wdst0 = &LW[0][wrow * 32 + whalf];
  _Float16* wdst1 = &LW[1][wrow * 32 + whalf];
  const _Float16* arow = Ah + (m0 + ln) * 1024;

#ifdef HAVE_ASYNC_LDS
  async_cp32B(wsrc, wdst0);
#else
  *(v16h*)wdst0 = *(const v16h*)(wsrc);
#endif
  v16h a = load_a_frag(arow, g);

  v8f acc[4] = {};
  int cur = 0;
  for (int k0 = 0; k0 < 1024; k0 += 32) {
    const bool more = (k0 + 32) < 1024;
    v16h anext = {};
#ifdef HAVE_ASYNC_LDS
    ASYNC_WAIT(0);
    __syncthreads();
    if (more) {
      async_cp32B(wsrc + k0 + 32, cur ? wdst0 : wdst1);
      anext = load_a_frag(arow + k0 + 32, g);
    }
#else
    __syncthreads();
    v16h wnext = {};
    if (more) {
      wnext = *(const v16h*)(wsrc + k0 + 32);
      anext = load_a_frag(arow + k0 + 32, g);
    }
#endif
    v16h bf[4];
#pragma unroll
    for (int nt = 0; nt < 4; nt++) {
      bf[nt] = *(const v16h*)&LW[cur][(nt * 16 + ln) * 32 + 16 * g];
    }
#pragma unroll
    for (int nt = 0; nt < 4; nt++) {
      acc[nt] = WMMA(a, bf[nt], acc[nt]);
    }
#ifndef HAVE_ASYNC_LDS
    if (more) *(v16h*)(cur ? wdst0 : wdst1) = wnext;
#endif
    if (more) a = anext;
    cur ^= 1;
  }

#pragma unroll
  for (int nt = 0; nt < 4; nt++) {
    const int n = n0 + nt * 16 + ln;
#pragma unroll
    for (int r = 0; r < 8; r++) {
      out[(m0 + r + 8 * g) * 1024 + n] = (float)acc[nt][r];
    }
  }
}

// ---------------------------------------------------------------------------
extern "C" void kernel_launch(void* const* d_in, const int* in_sizes, int n_in,
                              void* d_out, int out_size, void* d_ws, size_t ws_size,
                              hipStream_t stream) {
  (void)in_sizes; (void)n_in; (void)out_size; (void)ws_size;
  const float* x  = (const float*)d_in[0];
  const float* Wq = (const float*)d_in[1];
  const float* Wk = (const float*)d_in[2];
  const float* Wv = (const float*)d_in[3];
  const float* Wo = (const float*)d_in[4];
  float* out = (float*)d_out;

  const size_t NX = 4096u * 1024u;  // x / Q / K / V / attn-out element count
  const size_t NW = 1024u * 1024u;  // weight element count
  _Float16* xh  = (_Float16*)d_ws;
  _Float16* Wqh = xh + NX;
  _Float16* Wkh = Wqh + NW;
  _Float16* Wvh = Wkh + NW;
  _Float16* Woh = Wvh + NW;
  _Float16* Qh  = Woh + NW;   // [B,H,S,HD], pre-scaled by 1/8
  _Float16* Kh  = Qh + NX;    // [B,H,S,HD]
  _Float16* Vt  = Kh + NX;    // [B,H,HD,S] (transposed)
  _Float16* Ah  = Vt + NX;    // [B,S,E] concat-head attn output
  // total workspace: 24M halves = 48 MB

  f32_to_f16_kernel<<<dim3(NX / 1024), dim3(256), 0, stream>>>(x,  xh,  (int)NX);
  f32_to_f16_kernel<<<dim3(NW / 1024), dim3(256), 0, stream>>>(Wq, Wqh, (int)NW);
  f32_to_f16_kernel<<<dim3(NW / 1024), dim3(256), 0, stream>>>(Wk, Wkh, (int)NW);
  f32_to_f16_kernel<<<dim3(NW / 1024), dim3(256), 0, stream>>>(Wv, Wvh, (int)NW);
  f32_to_f16_kernel<<<dim3(NW / 1024), dim3(256), 0, stream>>>(Wo, Woh, (int)NW);

  qkv_proj_kernel<<<dim3(16, 64, 3), dim3(128), 0, stream>>>(
      xh, Wqh, Wkh, Wvh, Qh, Kh, Vt);
  flash_attn_kernel<<<dim3(32, 16, 2), dim3(128), 0, stream>>>(Qh, Kh, Vt, Ah);
  out_proj_kernel<<<dim3(16, 64), dim3(128), 0, stream>>>(Ah, Woh, out);
}
